// AttentionHead_90941637525879
// MI455X (gfx1250) — matrix operand order repaired and checked by reading, not compiled, not run on hardware
//
#include <hip/hip_runtime.h>
#include <hip/hip_bf16.h>
#include <stdint.h>

typedef _Float16 v16h __attribute__((ext_vector_type(16)));
typedef _Float16 h8   __attribute__((ext_vector_type(8)));
typedef float    v8f  __attribute__((ext_vector_type(8)));
typedef float    f8   __attribute__((ext_vector_type(8)));
typedef int      v4i  __attribute__((vector_size(16)));

#define EMBED 1024
#define HEAD  64
#define BATCH 4
#define SEQ   4096
#define BS    (BATCH*SEQ)
#define WAVES 4

// workspace layout (bytes), all offsets 256B aligned
#define WT_OFF 0
#define Q_OFF  (3*HEAD*EMBED*2)            // 393216
#define K_OFF  (Q_OFF + BS*HEAD*2)         // +2MB
#define VT_OFF (K_OFF + BS*HEAD*2)         // +2MB

// ---- gfx1250 async global->LDS path (guarded; falls back to plain staging) ----
#if defined(__has_builtin)
#if __has_builtin(__builtin_amdgcn_global_load_async_to_lds_b128) && \
    __has_builtin(__builtin_amdgcn_s_wait_asynccnt)
#define HAVE_ASYNC_LDS 1
#endif
#endif
#ifndef HAVE_ASYNC_LDS
#define HAVE_ASYNC_LDS 0
#endif

#if HAVE_ASYNC_LDS
#define AS1 __attribute__((address_space(1)))
#define AS3 __attribute__((address_space(3)))
#define WAIT_ASYNC(n) __builtin_amdgcn_s_wait_asynccnt(n)
#else
#define WAIT_ASYNC(n)
#endif

static __device__ inline v8f wmma_f16(v16h a, v16h b, v8f c) {
    return __builtin_amdgcn_wmma_f32_16x16x32_f16(false, a, false, b, (short)0, c,
                                                  false, false);
}

// ---------------------------------------------------------------------------
// Kernel 1: W [1024][64] fp32 -> W^T [64][1024] f16, for q/k/v
// ---------------------------------------------------------------------------
__global__ void prep_weights(const float* Wq, const float* Wk, const float* Wv,
                             _Float16* Wt) {
    int idx = blockIdx.x * blockDim.x + threadIdx.x;
    if (idx >= 3 * HEAD * EMBED) return;
    int w = idx / (HEAD * EMBED);
    int r = idx % (HEAD * EMBED);
    int h = r / EMBED;
    int k = r % EMBED;
    const float* W = (w == 0) ? Wq : ((w == 1) ? Wk : Wv);
    Wt[idx] = (_Float16)W[(size_t)k * HEAD + h];
}

// ---------------------------------------------------------------------------
// Kernel 2: QKV projection. One wave per (16-row tile, projection).
// A = x rows (fp32 -> f16 on the fly), B = W^T rows (contiguous 32B loads).
// ---------------------------------------------------------------------------
__global__ __launch_bounds__(32) void qkv_proj(const float* __restrict__ x,
                                               const _Float16* __restrict__ Wt,
                                               const float* __restrict__ bq,
                                               const float* __restrict__ bk,
                                               const float* __restrict__ bv,
                                               _Float16* __restrict__ Q,
                                               _Float16* __restrict__ K,
                                               _Float16* __restrict__ Vt) {
    const int tile  = blockIdx.x;        // 0 .. BS/16-1
    const int which = blockIdx.y;        // 0=q 1=k 2=v
    const int lane  = threadIdx.x & 31;
    const int n     = lane & 15;
    const int hi    = lane >> 4;
    const int abase = hi * 16;

    const _Float16* W    = Wt + (size_t)which * HEAD * EMBED;
    const float*    bias = (which == 0) ? bq : ((which == 1) ? bk : bv);

    const float* xp = x + (size_t)(tile * 16 + n) * EMBED;  // A row (m = lane%16)

    v8f acc[4] = {};
    for (int kk = 0; kk < EMBED; kk += 32) {
        // A fragment: k = abase+e, contiguous 16
        f8 lo = *(const f8*)(xp + kk + abase);
        f8 hh = *(const f8*)(xp + kk + abase + 8);
        v16h a;
#pragma unroll
        for (int i = 0; i < 8; ++i) {
            a[i]     = (_Float16)lo[i];
            a[i + 8] = (_Float16)hh[i];
        }
#pragma unroll
        for (int t = 0; t < 4; ++t) {
            // B fragment: k = hi*16 + e, contiguous 16 halves
            v16h b = *(const v16h*)(W + (size_t)(t * 16 + n) * EMBED + kk + hi * 16);
            acc[t] = wmma_f16(a, b, acc[t]);
        }
    }
    // C layout: acc[t][r] is (row = r + hi*8, col = t*16 + n); add bias, store f16
#pragma unroll
    for (int t = 0; t < 4; ++t) {
        float bval = bias[t * 16 + n];
        int   h    = t * 16 + n;
#pragma unroll
        for (int r = 0; r < 8; ++r) {
            int   srow = tile * 16 + r + hi * 8;
            float val  = acc[t][r] + bval;
            if (which == 0) {
                Q[(size_t)srow * HEAD + h] = (_Float16)val;
            } else if (which == 1) {
                K[(size_t)srow * HEAD + h] = (_Float16)val;
            } else {
                int bb = srow / SEQ, si = srow % SEQ;
                Vt[((size_t)bb * HEAD + h) * SEQ + si] = (_Float16)val;  // V^T
            }
        }
    }
}

// ---------------------------------------------------------------------------
// Cooperative staging of one 32-key step: K tile [32][64], V tile [64][32].
// 4 transfers of 16B per thread (128 threads -> 8KB). With async-LDS this is
// exactly 4 ASYNCcnt increments per wave.
// ---------------------------------------------------------------------------
__device__ __forceinline__ void stage_kv(const _Float16* __restrict__ Kb,
                                         const _Float16* __restrict__ Vb,
                                         int j0, _Float16* kt, _Float16* vt,
                                         int tid) {
    const int krow = tid >> 3, koff = (tid & 7) * 8;   // 16 rows x 8 chunks
    const int vrow = tid >> 2, voff = (tid & 3) * 8;   // 32 rows x 4 chunks
    const _Float16* kg0 = Kb + (size_t)(j0 + krow) * HEAD + koff;
    const _Float16* kg1 = Kb + (size_t)(j0 + 16 + krow) * HEAD + koff;
    const _Float16* vg0 = Vb + (size_t)vrow * SEQ + j0 + voff;
    const _Float16* vg1 = Vb + (size_t)(32 + vrow) * SEQ + j0 + voff;
    _Float16* kl0 = kt + krow * 64 + koff;
    _Float16* kl1 = kt + (16 + krow) * 64 + koff;
    _Float16* vl0 = vt + vrow * 32 + voff;
    _Float16* vl1 = vt + (32 + vrow) * 32 + voff;
#if HAVE_ASYNC_LDS
    __builtin_amdgcn_global_load_async_to_lds_b128((AS1 v4i*)kg0, (AS3 v4i*)kl0, 0, 0);
    __builtin_amdgcn_global_load_async_to_lds_b128((AS1 v4i*)kg1, (AS3 v4i*)kl1, 0, 0);
    __builtin_amdgcn_global_load_async_to_lds_b128((AS1 v4i*)vg0, (AS3 v4i*)vl0, 0, 0);
    __builtin_amdgcn_global_load_async_to_lds_b128((AS1 v4i*)vg1, (AS3 v4i*)vl1, 0, 0);
#else
    *(h8*)kl0 = *(const h8*)kg0;
    *(h8*)kl1 = *(const h8*)kg1;
    *(h8*)vl0 = *(const h8*)vg0;
    *(h8*)vl1 = *(const h8*)vg1;
#endif
}

// ---------------------------------------------------------------------------
// Kernel 3: flash attention. 4 waves/block (64 q rows, one batch).
// K/V tiles double-buffered in LDS via async global->LDS DMA; per 32-key step:
// 4 WMMA (QK^T) + online softmax + P via LDS + 4 WMMA (PV).
// ---------------------------------------------------------------------------
__global__ __launch_bounds__(WAVES * 32) void attn(const _Float16* __restrict__ Q,
                                                   const _Float16* __restrict__ K,
                                                   const _Float16* __restrict__ Vt,
                                                   float* __restrict__ out) {
    __shared__ __align__(64) _Float16 Ktile[2][32][64];   // [buf][key][head]
    __shared__ __align__(64) _Float16 Vtile[2][64][32];   // [buf][head][key]
    __shared__ __align__(64) _Float16 Ptile[WAVES][16][32];

    const int tid   = threadIdx.x;
    const int wave  = tid >> 5;
    const int lane  = tid & 31;
    const int n     = lane & 15;
    const int hi    = lane >> 4;
    const int abase = hi * 16;

    const int qtile = blockIdx.x * WAVES + wave;  // 16-row tile over B*S
    const int g0    = qtile * 16;
    const int b     = g0 / SEQ;
    const _Float16* Kb = K + (size_t)b * SEQ * HEAD;
    const _Float16* Vb = Vt + (size_t)b * HEAD * SEQ;

    // Q fragments (A operand), head dim = two K-chunks of 32
    const _Float16* qrow = Q + (size_t)(g0 + n) * HEAD;
    v16h qa0, qa1;
    {
        h8 l0 = *(const h8*)(qrow + 0 + abase);
        h8 h0 = *(const h8*)(qrow + 0 + abase + 8);
        h8 l1 = *(const h8*)(qrow + 32 + abase);
        h8 h1 = *(const h8*)(qrow + 32 + abase + 8);
#pragma unroll
        for (int i = 0; i < 8; ++i) {
            qa0[i] = l0[i]; qa0[i + 8] = h0[i];
            qa1[i] = l1[i]; qa1[i + 8] = h1[i];
        }
    }

    const float c = 0.125f * 1.44269504088896340736f;  // log2(e)/sqrt(HEAD)

    v8f   o[4] = {};
    float mrow[8], lrow[8];
#pragma unroll
    for (int r = 0; r < 8; ++r) { mrow[r] = -3.0e38f; lrow[r] = 0.0f; }

    const int NITER = SEQ / 32;
    stage_kv(Kb, Vb, 0, &Ktile[0][0][0], &Vtile[0][0][0], tid);

    for (int i = 0; i < NITER; ++i) {
        const int buf = i & 1;
        if (i + 1 < NITER) {
            stage_kv(Kb, Vb, (i + 1) * 32, &Ktile[buf ^ 1][0][0],
                     &Vtile[buf ^ 1][0][0], tid);
            WAIT_ASYNC(4);   // 4 newest = prefetch; current buffer complete
        } else {
            WAIT_ASYNC(0);
        }
        __syncthreads();

        // ---- scores: two 16x16 tiles over 32 keys (K from LDS) ----
        v8f s0 = {}, s1 = {};
        {
            const _Float16* k0p = &Ktile[buf][n][0];
            const _Float16* k1p = &Ktile[buf][16 + n][0];
            v16h b00 = *(const v16h*)(k0p + 0 + hi * 16);
            v16h b01 = *(const v16h*)(k0p + 32 + hi * 16);
            v16h b10 = *(const v16h*)(k1p + 0 + hi * 16);
            v16h b11 = *(const v16h*)(k1p + 32 + hi * 16);
            s0 = wmma_f16(qa0, b00, s0);
            s0 = wmma_f16(qa1, b01, s0);
            s1 = wmma_f16(qa0, b10, s1);
            s1 = wmma_f16(qa1, b11, s1);
        }

        // ---- online softmax across the 32 columns ----
#pragma unroll
        for (int r = 0; r < 8; ++r) {
            float v0 = s0[r], v1 = s1[r];
            float mx = fmaxf(v0, v1);
#pragma unroll
            for (int d = 1; d < 16; d <<= 1) mx = fmaxf(mx, __shfl_xor(mx, d, 32));
            float mnew  = fmaxf(mrow[r], mx);
            float alpha = exp2f((mrow[r] - mnew) * c);
            float p0    = exp2f((v0 - mnew) * c);
            float p1    = exp2f((v1 - mnew) * c);
            float ps    = p0 + p1;
#pragma unroll
            for (int d = 1; d < 16; d <<= 1) ps += __shfl_xor(ps, d, 32);
            lrow[r] = lrow[r] * alpha + ps;
            mrow[r] = mnew;
            o[0][r] *= alpha; o[1][r] *= alpha; o[2][r] *= alpha; o[3][r] *= alpha;
            int mr = r + hi * 8;
            Ptile[wave][mr][n]      = (_Float16)p0;
            Ptile[wave][mr][16 + n] = (_Float16)p1;
        }

        // ---- reload P in A-operand layout (wave-private LDS region) ----
        v16h pa;
        {
            const _Float16* prow = &Ptile[wave][n][0];
            h8 lo = *(const h8*)(prow + abase);
            h8 hh = *(const h8*)(prow + abase + 8);
#pragma unroll
            for (int ii = 0; ii < 8; ++ii) { pa[ii] = lo[ii]; pa[ii + 8] = hh[ii]; }
        }

        // ---- PV: o[t] += P(16x32) @ V[32 x 16] (V from LDS) ----
#pragma unroll
        for (int t = 0; t < 4; ++t) {
            v16h bfrag = *(const v16h*)(&Vtile[buf][t * 16 + n][hi * 16]);
            o[t] = wmma_f16(pa, bfrag, o[t]);
        }
        __syncthreads();
    }

    // ---- finalize: divide by l, store fp32 ----
#pragma unroll
    for (int r = 0; r < 8; ++r) {
        float inv  = 1.0f / lrow[r];
        int   srow = g0 + r + hi * 8;
#pragma unroll
        for (int t = 0; t < 4; ++t)
            out[(size_t)srow * HEAD + t * 16 + n] = o[t][r] * inv;
    }
}

// ---------------------------------------------------------------------------
extern "C" void kernel_launch(void* const* d_in, const int* in_sizes, int n_in,
                              void* d_out, int out_size, void* d_ws, size_t ws_size,
                              hipStream_t stream) {
    const float* x  = (const float*)d_in[0];
    const float* Wq = (const float*)d_in[1];
    const float* bq = (const float*)d_in[2];
    const float* Wk = (const float*)d_in[3];
    const float* bk = (const float*)d_in[4];
    const float* Wv = (const float*)d_in[5];
    const float* bv = (const float*)d_in[6];
    float* out = (float*)d_out;

    char* ws = (char*)d_ws;
    _Float16* Wt = (_Float16*)(ws + WT_OFF);
    _Float16* Q  = (_Float16*)(ws + Q_OFF);
    _Float16* K  = (_Float16*)(ws + K_OFF);
    _Float16* Vt = (_Float16*)(ws + VT_OFF);

    prep_weights<<<(3 * HEAD * EMBED + 255) / 256, 256, 0, stream>>>(Wq, Wk, Wv, Wt);

    dim3 pgrid(BS / 16, 3);
    qkv_proj<<<pgrid, 32, 0, stream>>>(x, Wt, bq, bk, bv, Q, K, Vt);

    attn<<<BS / (16 * WAVES), WAVES * 32, 0, stream>>>(Q, K, Vt, out);
}
